// Attention_91104846283177
// MI455X (gfx1250) — compile-verified, matrix operand order checked by reading, not run on hardware
//
#include <hip/hip_runtime.h>

// ---------------------------------------------------------------------------
// Types for CDNA5 WMMA (wave32, 16x16x32 bf16 -> f32)
// ---------------------------------------------------------------------------
typedef __attribute__((ext_vector_type(16))) __bf16          v16bf;
typedef __attribute__((ext_vector_type(8)))  float           v8f;
typedef __attribute__((ext_vector_type(8)))  unsigned short  v8u;
typedef __attribute__((ext_vector_type(16))) unsigned short  v16u;
typedef __attribute__((ext_vector_type(4)))  int             v4i;

#define HEADS    16
#define DIMH     64
#define SEQ      2048
#define BATCH    2
#define DMODEL   1024
#define MTOT     (BATCH * SEQ)      // 4096 rows
#define NEG_INF  (-3.0e38f)

// ---------------------------------------------------------------------------
// Async global->LDS (CDNA5 GLOBAL_LOAD_ASYNC_TO_LDS_B128), guarded so the
// file still compiles on toolchains without the builtin.
// ---------------------------------------------------------------------------
#if defined(__has_builtin)
#if __has_builtin(__builtin_amdgcn_global_load_async_to_lds_b128) && \
    __has_builtin(__builtin_amdgcn_s_wait_asynccnt)
#define USE_ASYNC_LDS 1
#endif
#endif
#ifndef USE_ASYNC_LDS
#define USE_ASYNC_LDS 0
#endif

#define AS1 __attribute__((address_space(1)))
#define AS3 __attribute__((address_space(3)))

// copy 16 bytes global -> LDS (async when available)
__device__ __forceinline__ void cp16(const unsigned short* g, unsigned short* l) {
#if USE_ASYNC_LDS
    // flat->AS1: same 64-bit address; flat->AS3: low 32 bits are the LDS offset
    __builtin_amdgcn_global_load_async_to_lds_b128(
        (AS1 v4i*)(unsigned long long)g,
        (AS3 v4i*)(unsigned)(unsigned long long)l,
        0, 0);
#else
    *(v8u*)l = *(const v8u*)g;
#endif
}

__device__ __forceinline__ void async_wait_all() {
#if USE_ASYNC_LDS
    __builtin_amdgcn_s_wait_asynccnt(0);
#endif
}

// float -> bf16 (round to nearest even), pure bit math
__device__ __forceinline__ unsigned short f2bf(float f) {
    unsigned int u = __float_as_uint(f);
    u += 0x7fffu + ((u >> 16) & 1u);
    return (unsigned short)(u >> 16);
}

// Load one 16x32 bf16 WMMA fragment (A or B role — same per-lane addressing):
// lane&15 selects the entity (row of A / col of B), lanes>=16 take K+8 / K+24.
__device__ __forceinline__ v16bf load_frag(const unsigned short* p, int ld) {
    const int lane = threadIdx.x & 31;
    const unsigned short* q = p + (long)(lane & 15) * ld + ((lane >> 4) << 3);
    v8u lo = *(const v8u*)(q);        // K = base+0..7
    v8u hi = *(const v8u*)(q + 16);   // K = base+16..23
    v16u u;
#pragma unroll
    for (int i = 0; i < 8; ++i) { u[i] = lo[i]; u[i + 8] = hi[i]; }
    return __builtin_bit_cast(v16bf, u);
}

__device__ __forceinline__ v8f wmma_bf16(v16bf a, v16bf b, v8f c) {
    return __builtin_amdgcn_wmma_f32_16x16x32_bf16(
        false, a, false, b, (short)0, c, false, false);
}

// ---------------------------------------------------------------------------
// Conversion kernels (fp32 -> bf16; weights stored transposed: Wt[n][k])
// ---------------------------------------------------------------------------
__global__ void cvt_tokens_kernel(const float* __restrict__ in,
                                  unsigned short* __restrict__ out) {
    int i = blockIdx.x * blockDim.x + threadIdx.x;
    out[i] = f2bf(in[i]);
}

__global__ void cvt_wT_kernel(const float* __restrict__ in,
                              unsigned short* __restrict__ out,
                              int ldin, int col0) {
    int idx = blockIdx.x * blockDim.x + threadIdx.x;
    int k = idx & (DMODEL - 1);
    int n = idx >> 10;
    out[(long)n * DMODEL + k] = f2bf(in[(long)k * ldin + col0 + n]);
}

// ---------------------------------------------------------------------------
// 128x128 block GEMM core: 256 threads = 8 waves, wave tile 32x64.
// Double-buffered LDS staging of A(128x32) and Bt(128x32) per K-step.
// ---------------------------------------------------------------------------
__device__ __forceinline__ void stage_ab(const unsigned short* Ag,
                                         const unsigned short* Bg, int kb,
                                         unsigned short* la, unsigned short* lb) {
    int t = threadIdx.x;
#pragma unroll
    for (int it = 0; it < 2; ++it) {
        int c = t + it * 256;              // 512 chunks of 16B per matrix
        int row = c >> 2, seg = (c & 3) << 3;
        cp16(Ag + (long)row * DMODEL + kb + seg, la + row * 32 + seg);
        cp16(Bg + (long)row * DMODEL + kb + seg, lb + row * 32 + seg);
    }
}

__device__ __forceinline__ void gemm_block_128x128(
        const unsigned short* Ag, const unsigned short* Bg,
        unsigned short* lA0, unsigned short* lA1,
        unsigned short* lB0, unsigned short* lB1,
        v8f acc[2][4], int wm, int wn) {
    stage_ab(Ag, Bg, 0, lA0, lB0);
    for (int kb = 0; kb < DMODEL; kb += 32) {
        unsigned short* la  = (kb & 32) ? lA1 : lA0;
        unsigned short* lb  = (kb & 32) ? lB1 : lB0;
        unsigned short* lan = (kb & 32) ? lA0 : lA1;
        unsigned short* lbn = (kb & 32) ? lB0 : lB1;
        async_wait_all();
        __syncthreads();                       // staged data visible to all
        if (kb + 32 < DMODEL) stage_ab(Ag, Bg, kb + 32, lan, lbn);
        v16bf a0 = load_frag(la + (wm * 32 + 0)  * 32, 32);
        v16bf a1 = load_frag(la + (wm * 32 + 16) * 32, 32);
#pragma unroll
        for (int t = 0; t < 4; ++t) {
            v16bf bf = load_frag(lb + (wn * 64 + t * 16) * 32, 32);
            acc[0][t] = wmma_bf16(a0, bf, acc[0][t]);
            acc[1][t] = wmma_bf16(a1, bf, acc[1][t]);
        }
        __syncthreads();                       // reads done before overwrite
    }
}

// ---------------------------------------------------------------------------
// QKV projection: C = tokens_bf(4096x1024) * Wt^T, scatter into head layout.
// vlayout==0: out[((b*16+h)*2048 + n)*64 + d]   (Q scaled, and K)
// vlayout==1: out[((b*16+h)*64 + d)*2048 + n]   (V transposed)
// ---------------------------------------------------------------------------
__global__ __launch_bounds__(256) void proj_qkv_kernel(
        const unsigned short* __restrict__ Abf,
        const unsigned short* __restrict__ Wt,
        unsigned short* __restrict__ out,
        int vlayout, float scale) {
    __shared__ alignas(16) unsigned short lA[2][128 * 32];
    __shared__ alignas(16) unsigned short lB[2][128 * 32];
    int bn = blockIdx.x & 7, bm = blockIdx.x >> 3;      // N/128=8, M/128=32
    int wv = threadIdx.x >> 5, lane = threadIdx.x & 31;
    int wm = wv >> 1, wn = wv & 1;

    v8f acc[2][4] = {};
    gemm_block_128x128(Abf + (long)bm * 128 * DMODEL, Wt + (long)bn * 128 * DMODEL,
                       lA[0], lA[1], lB[0], lB[1], acc, wm, wn);

    int cloc = lane & 15, hi8 = (lane >> 4) << 3;
#pragma unroll
    for (int i = 0; i < 2; ++i) {
#pragma unroll
        for (int t = 0; t < 4; ++t) {
            int c = bn * 128 + wn * 64 + t * 16 + cloc;
            int h = c >> 6, d = c & 63;
#pragma unroll
            for (int r = 0; r < 8; ++r) {
                int m = bm * 128 + wm * 32 + i * 16 + hi8 + r;
                int b = m >> 11, n = m & (SEQ - 1);
                float v = acc[i][t][r] * scale;
                long idx = vlayout
                    ? ((long)((b * HEADS + h) * DIMH + d)) * SEQ + n
                    : ((long)((b * HEADS + h) * SEQ + n)) * DIMH + d;
                out[idx] = f2bf(v);
            }
        }
    }
}

// ---------------------------------------------------------------------------
// Flash attention: block = 128 threads (4 waves) = 64 queries of one (b,h).
// K(32x64) and V^T(64x32) tiles double-buffered through LDS, shared by the
// 4 waves. S^T = K*Q^T so softmax is 8 local ops + one xor-16 shuffle, and
// the S^T accumulator repacks in-register into the P fragment for P*V.
// ---------------------------------------------------------------------------
__device__ __forceinline__ void stage_kv(const unsigned short* Kg,
                                         const unsigned short* Vg, int jb,
                                         unsigned short* lk, unsigned short* lv) {
    int t = threadIdx.x;
#pragma unroll
    for (int it = 0; it < 2; ++it) {
        int c = t + it * 128;                   // 256 chunks each
        int kr = c >> 3, ks = (c & 7) << 3;     // K: 32 rows x 64
        cp16(Kg + (long)(jb + kr) * DIMH + ks, lk + kr * DIMH + ks);
        int vr = c >> 2, vs = (c & 3) << 3;     // V: 64 rows x 32
        cp16(Vg + (long)vr * SEQ + jb + vs, lv + vr * 32 + vs);
    }
}

__global__ __launch_bounds__(128) void attn_kernel(
        const unsigned short* __restrict__ Qh,
        const unsigned short* __restrict__ Kh,
        const unsigned short* __restrict__ Vt,
        const unsigned char*  __restrict__ mask,
        unsigned short* __restrict__ Ah) {
    __shared__ alignas(16) unsigned short lK[2][32 * 64];
    __shared__ alignas(16) unsigned short lV[2][64 * 32];
    int bh = blockIdx.x >> 5;                  // 0..31 (b*16+h)
    int qb = blockIdx.x & 31;                  // 0..31 query block of 64
    int b = bh >> 4, h = bh & 15;
    int wv = threadIdx.x >> 5, lane = threadIdx.x & 31;
    int i0 = qb * 64 + wv * 16;
    int hi = lane >> 4;

    const unsigned short* Kbh = Kh + (long)bh * SEQ * DIMH;
    const unsigned short* Vbh = Vt + (long)bh * DIMH * SEQ;
    const unsigned char*  mrow = mask + (long)b * SEQ;

    // Q as B-fragments (d-chunks 0..31 and 32..63), loaded once from global
    const unsigned short* Qbase = Qh + ((long)bh * SEQ + i0) * DIMH;
    v16bf bq0 = load_frag(Qbase + 0,  DIMH);
    v16bf bq1 = load_frag(Qbase + 32, DIMH);

    v8f o[4] = {};
    float m_i = NEG_INF, l_i = 0.f;

    stage_kv(Kbh, Vbh, 0, lK[0], lV[0]);
    for (int jb = 0; jb < SEQ; jb += 32) {
        int cur = (jb >> 5) & 1;
        async_wait_all();
        __syncthreads();
        if (jb + 32 < SEQ) stage_kv(Kbh, Vbh, jb + 32, lK[cur ^ 1], lV[cur ^ 1]);

        const unsigned short* k0 = lK[cur];
        const unsigned short* k1 = lK[cur] + 16 * DIMH;
        v8f st0 = {}, st1 = {};
        st0 = wmma_bf16(load_frag(k0,      DIMH), bq0, st0);
        st0 = wmma_bf16(load_frag(k0 + 32, DIMH), bq1, st0);
        st1 = wmma_bf16(load_frag(k1,      DIMH), bq0, st1);
        st1 = wmma_bf16(load_frag(k1 + 32, DIMH), bq1, st1);

        unsigned long long mk0 = *(const unsigned long long*)(mrow + jb + hi * 8);
        unsigned long long mk1 = *(const unsigned long long*)(mrow + jb + 16 + hi * 8);

        float s0[8], s1[8], mloc = NEG_INF;
#pragma unroll
        for (int r = 0; r < 8; ++r) {
            s0[r] = ((mk0 >> (8 * r)) & 0xffull) ? st0[r] : NEG_INF;
            s1[r] = ((mk1 >> (8 * r)) & 0xffull) ? st1[r] : NEG_INF;
            mloc = fmaxf(mloc, fmaxf(s0[r], s1[r]));
        }
        mloc = fmaxf(mloc, __shfl_xor(mloc, 16, 32));
        float mnew  = fmaxf(m_i, mloc);
        float alpha = __expf(m_i - mnew);

        float p0[8], p1[8], ls = 0.f;
#pragma unroll
        for (int r = 0; r < 8; ++r) {
            p0[r] = __expf(s0[r] - mnew);
            p1[r] = __expf(s1[r] - mnew);
            ls += p0[r] + p1[r];
        }
        ls += __shfl_xor(ls, 16, 32);
        l_i = l_i * alpha + ls;
        m_i = mnew;

        float ar[8];
#pragma unroll
        for (int r = 0; r < 8; ++r) ar[r] = __shfl(alpha, (hi << 3) + r, 32);
#pragma unroll
        for (int t = 0; t < 4; ++t)
#pragma unroll
            for (int r = 0; r < 8; ++r) o[t][r] *= ar[r];

        v16u pu;
#pragma unroll
        for (int r = 0; r < 8; ++r) { pu[r] = f2bf(p0[r]); pu[r + 8] = f2bf(p1[r]); }
        v16bf ap = __builtin_bit_cast(v16bf, pu);

#pragma unroll
        for (int t = 0; t < 4; ++t)
            o[t] = wmma_bf16(ap, load_frag(lV[cur] + t * 16 * 32, 32), o[t]);

        __syncthreads();
    }

    float inv = 1.0f / l_i;
    float ir[8];
#pragma unroll
    for (int r = 0; r < 8; ++r) ir[r] = __shfl(inv, (hi << 3) + r, 32);

#pragma unroll
    for (int t = 0; t < 4; ++t) {
        int dg = h * DIMH + t * 16 + (lane & 15);
#pragma unroll
        for (int r = 0; r < 8; ++r) {
            int m = i0 + (hi << 3) + r;
            Ah[((long)(b * SEQ + m)) * (HEADS * DIMH) + dg] = f2bf(o[t][r] * ir[r]);
        }
    }
}

// ---------------------------------------------------------------------------
// Output projection: d_out(4096x1024 fp32) = Ah * WoT^T
// ---------------------------------------------------------------------------
__global__ __launch_bounds__(256) void proj_out_kernel(
        const unsigned short* __restrict__ Abf,
        const unsigned short* __restrict__ Wt,
        float* __restrict__ out) {
    __shared__ alignas(16) unsigned short lA[2][128 * 32];
    __shared__ alignas(16) unsigned short lB[2][128 * 32];
    int bn = blockIdx.x & 7, bm = blockIdx.x >> 3;
    int wv = threadIdx.x >> 5, lane = threadIdx.x & 31;
    int wm = wv >> 1, wn = wv & 1;

    v8f acc[2][4] = {};
    gemm_block_128x128(Abf + (long)bm * 128 * DMODEL, Wt + (long)bn * 128 * DMODEL,
                       lA[0], lA[1], lB[0], lB[1], acc, wm, wn);

    int cloc = lane & 15, hi8 = (lane >> 4) << 3;
#pragma unroll
    for (int i = 0; i < 2; ++i) {
#pragma unroll
        for (int t = 0; t < 4; ++t) {
            int c = bn * 128 + wn * 64 + t * 16 + cloc;
#pragma unroll
            for (int r = 0; r < 8; ++r) {
                int m = bm * 128 + wm * 32 + i * 16 + hi8 + r;
                out[(long)m * DMODEL + c] = acc[i][t][r];
            }
        }
    }
}

// ---------------------------------------------------------------------------
// Launch
// ---------------------------------------------------------------------------
extern "C" void kernel_launch(void* const* d_in, const int* in_sizes, int n_in,
                              void* d_out, int out_size, void* d_ws, size_t ws_size,
                              hipStream_t stream) {
    (void)in_sizes; (void)n_in; (void)out_size; (void)ws_size;
    const float*         tokens = (const float*)d_in[0];
    const unsigned char* cmask  = (const unsigned char*)d_in[1];
    const float*         Wq     = (const float*)d_in[2];
    const float*         Wkv    = (const float*)d_in[3];
    const float*         Wo     = (const float*)d_in[4];
    float*               out    = (float*)d_out;

    char* ws = (char*)d_ws;
    const size_t MB = 1024 * 1024;
    unsigned short* tok_bf = (unsigned short*)(ws);                // 8 MiB
    unsigned short* WqT    = (unsigned short*)(ws + 8  * MB);      // 2 MiB
    unsigned short* WkT    = (unsigned short*)(ws + 10 * MB);      // 2 MiB
    unsigned short* WvT    = (unsigned short*)(ws + 12 * MB);      // 2 MiB
    unsigned short* WoT    = (unsigned short*)(ws + 14 * MB);      // 2 MiB
    unsigned short* Qh     = (unsigned short*)(ws + 16 * MB);      // 8 MiB
    unsigned short* Kh     = (unsigned short*)(ws + 24 * MB);      // 8 MiB
    unsigned short* Vt     = (unsigned short*)(ws + 32 * MB);      // 8 MiB
    unsigned short* Ah     = (unsigned short*)(ws + 40 * MB);      // 8 MiB

    cvt_tokens_kernel<<<(MTOT * DMODEL) / 256, 256, 0, stream>>>(tokens, tok_bf);
    cvt_wT_kernel<<<(DMODEL * DMODEL) / 256, 256, 0, stream>>>(Wq,  WqT, DMODEL,     0);
    cvt_wT_kernel<<<(DMODEL * DMODEL) / 256, 256, 0, stream>>>(Wkv, WkT, 2 * DMODEL, 0);
    cvt_wT_kernel<<<(DMODEL * DMODEL) / 256, 256, 0, stream>>>(Wkv, WvT, 2 * DMODEL, DMODEL);
    cvt_wT_kernel<<<(DMODEL * DMODEL) / 256, 256, 0, stream>>>(Wo,  WoT, DMODEL,     0);

    proj_qkv_kernel<<<256, 256, 0, stream>>>(tok_bf, WqT, Qh, 0, 0.125f);
    proj_qkv_kernel<<<256, 256, 0, stream>>>(tok_bf, WkT, Kh, 0, 1.0f);
    proj_qkv_kernel<<<256, 256, 0, stream>>>(tok_bf, WvT, Vt, 1, 1.0f);

    attn_kernel<<<1024, 128, 0, stream>>>(Qh, Kh, Vt, cmask, Ah);

    proj_out_kernel<<<256, 256, 0, stream>>>(Ah, WoT, out);
}
